// LeNet_5_73744588472323
// MI455X (gfx1250) — compile-verified, matrix operand order checked by reading, not run on hardware
//
#include <hip/hip_runtime.h>

typedef __attribute__((ext_vector_type(16))) _Float16 v16h;
typedef __attribute__((ext_vector_type(8)))  float    v8f;

// ---------------------------------------------------------------------------
// Pack f32 -> f16 (used once for the network input x).
// ---------------------------------------------------------------------------
__global__ __launch_bounds__(256) void pack_f32_to_f16(const float* __restrict__ src,
                                                       _Float16* __restrict__ dst, int n) {
    int i = blockIdx.x * 256 + threadIdx.x;
    if (i < n) dst[i] = (_Float16)src[i];
}

// ---------------------------------------------------------------------------
// Weight prep: f32 [O, K] (flat conv weight, unfold-compatible ordering)
//  -> f16 column-major padded [Npad][Kpad], zero-filled padding.
// Zero padding is what makes the GEMM K-loop guard-free.
// ---------------------------------------------------------------------------
__global__ __launch_bounds__(256) void prep_weights(const float* __restrict__ w,
                                                    _Float16* __restrict__ wB,
                                                    int Kdim, int Kpad, int O, int Npad) {
    int i = blockIdx.x * 256 + threadIdx.x;
    if (i >= Npad * Kpad) return;
    int n = i / Kpad;
    int k = i - n * Kpad;
    _Float16 v = (_Float16)0.0f;
    if (n < O && k < Kdim) v = (_Float16)w[n * Kdim + k];
    wB[i] = v;
}

// ---------------------------------------------------------------------------
// A-fragment gather. 16-bit A layout (ISA 7.12.2):
//   lane L holds row M = L%16; half = L/16 selects K sub-block;
//   element e -> v = e/2, j = e&1;  K = (v/4)*16 + half*8 + (v%4)*2 + j
// `in` stays an SGPR base; rowOff+koff is a 32-bit VGPR index so loads lower
// to global_load_u16 vOff, s[base] scale_offset (GVS mode, no 64-bit math).
// koff runs are 32B-aligned runs of 8 ints -> ds_load_b128.
// ---------------------------------------------------------------------------
__device__ inline v16h load_a_frag(const _Float16* __restrict__ in, int rowOff,
                                   const int* __restrict__ koffLDS, int kb, int half) {
    v16h a;
#pragma unroll
    for (int e = 0; e < 16; ++e) {
        const int v = e >> 1, j = e & 1;
        const int k = kb + ((v >> 2) << 4) + (half << 3) + ((v & 3) << 1) + j;
        a[e] = in[rowOff + koffLDS[k]];
    }
    return a;
}

// B-fragment from LDS column n (contiguous 8-element K runs -> ds_load_b128)
template<int Kpad>
__device__ inline v16h load_b_frag(const _Float16* __restrict__ wcol, int kb, int half) {
    v16h bf;
#pragma unroll
    for (int e = 0; e < 16; ++e) {
        const int v = e >> 1, j = e & 1;
        const int k = kb + ((v >> 2) << 4) + (half << 3) + ((v & 3) << 1) + j;
        bf[e] = wcol[k];
    }
    return bf;
}

// ---------------------------------------------------------------------------
// Fused implicit-GEMM conv (+bias, +ReLU) via V_WMMA_F32_16X16X32_F16.
// Block = 256 threads = 8 waves; each wave owns a 32-row M tile (MT=2
// sub-tiles of 16) x Npad columns. Weights + im2col offset table in LDS.
// Inner loop is branch-free: padded K reads in[rowOff+0] * 0-weight.
// ---------------------------------------------------------------------------
template<typename OutT, int Cin, int Hin, int Win, int KH, int KW,
         int O, int Ho, int Wo, int Kdim, int Kpad, int Npad, bool RELU>
__global__ __launch_bounds__(256) void conv_gemm_wmma(
    const _Float16* __restrict__ in, const _Float16* __restrict__ wB,
    const float* __restrict__ bias, OutT* __restrict__ out, int Mtotal)
{
    __shared__ alignas(16) _Float16 wLDS[Npad * Kpad];
    __shared__ alignas(16) int      koffLDS[Kpad];

    // Stage weights (16B copies) and build the im2col offset table.
    {
        const uint4* src = (const uint4*)wB;
        uint4* dst = (uint4*)wLDS;
        constexpr int nVec = (Npad * Kpad * 2) / 16;
        for (int i = threadIdx.x; i < nVec; i += 256) dst[i] = src[i];

        for (int k = threadIdx.x; k < Kpad; k += 256) {
            int offv = 0;
            if (k < Kdim) {
                const int c  = k / (KH * KW);
                const int r  = k - c * (KH * KW);
                const int ky = r / KW;
                const int kx = r - ky * KW;
                offv = (c * Hin + ky) * Win + kx;
            }
            koffLDS[k] = offv;
        }
    }
    __syncthreads();

    const int lane  = threadIdx.x & 31;
    const int wave  = threadIdx.x >> 5;
    const int mBase = (blockIdx.x * 8 + wave) * 32;   // 32 rows per wave (MT=2)
    if (mBase >= Mtotal) return;   // wave-uniform; EXEC stays all-ones

    const int half = lane >> 4;
    const int nIn  = lane & 15;
    constexpr int MT        = 2;
    constexpr int NT        = Npad / 16;
    constexpr int LHW       = Ho * Wo;
    constexpr int ImgStride = Cin * Hin * Win;

    // Rows owned by this lane for A loads (clamped; garbage rows masked at store)
    int rowOff[MT];
#pragma unroll
    for (int i = 0; i < MT; ++i) {
        int m  = mBase + i * 16 + (lane & 15);
        int mc = (m < Mtotal) ? m : (Mtotal - 1);
        const int bImg = mc / LHW;
        const int l    = mc - bImg * LHW;
        const int oy   = l / Wo;
        const int ox   = l - oy * Wo;
        rowOff[i] = bImg * ImgStride + oy * Win + ox;
    }

    const v8f zero = {0.f, 0.f, 0.f, 0.f, 0.f, 0.f, 0.f, 0.f};
    v8f acc[MT][NT];
#pragma unroll
    for (int i = 0; i < MT; ++i)
#pragma unroll
        for (int t = 0; t < NT; ++t) acc[i][t] = zero;

    for (int kb = 0; kb < Kpad; kb += 32) {
        v16h a0 = load_a_frag(in, rowOff[0], koffLDS, kb, half);
        v16h a1 = load_a_frag(in, rowOff[1], koffLDS, kb, half);
#pragma unroll
        for (int t = 0; t < NT; ++t) {
            v16h bfrag = load_b_frag<Kpad>(&wLDS[(t * 16 + nIn) * Kpad], kb, half);
            acc[0][t] = __builtin_amdgcn_wmma_f32_16x16x32_f16(
                false, a0, false, bfrag, (short)0, acc[0][t], false, false);
            acc[1][t] = __builtin_amdgcn_wmma_f32_16x16x32_f16(
                false, a1, false, bfrag, (short)0, acc[1][t], false, false);
        }
    }

    // Epilogue: D layout (ISA 7.12.2): VGPR r at lane L -> M = r + (L/16)*8, N = L%16
#pragma unroll
    for (int i = 0; i < MT; ++i) {
#pragma unroll
        for (int t = 0; t < NT; ++t) {
            const int n = t * 16 + nIn;
#pragma unroll
            for (int r = 0; r < 8; ++r) {
                const int mm = mBase + i * 16 + r + half * 8;
                if (n < O && mm < Mtotal) {
                    const int bb = mm / LHW;
                    const int ll = mm - bb * LHW;
                    float v = acc[i][t][r] + bias[n];
                    if (RELU) v = fmaxf(v, 0.0f);
                    out[(bb * O + n) * LHW + ll] = (OutT)v;
                }
            }
        }
    }
}

// ---------------------------------------------------------------------------
// Host side: input pack + 4 weight preps + 4 fused conv/GEMM launches on
// `stream`. Intermediates kept as f16 in workspace (~59 MB).
// ---------------------------------------------------------------------------
extern "C" void kernel_launch(void* const* d_in, const int* in_sizes, int n_in,
                              void* d_out, int out_size, void* d_ws, size_t ws_size,
                              hipStream_t stream) {
    (void)in_sizes; (void)n_in; (void)out_size; (void)ws_size;

    const float* x  = (const float*)d_in[0];
    const float* w1 = (const float*)d_in[1];
    const float* b1 = (const float*)d_in[2];
    const float* w2 = (const float*)d_in[3];
    const float* b2 = (const float*)d_in[4];
    const float* w3 = (const float*)d_in[5];
    const float* b3 = (const float*)d_in[6];
    const float* wf = (const float*)d_in[7];
    const float* bf = (const float*)d_in[8];
    float* out = (float*)d_out;

    char* ws = (char*)d_ws;
    size_t off = 0;
    auto carve = [&](size_t bytes) -> char* {
        char* p = ws + off;
        off = (off + bytes + 255) & ~(size_t)255;
        return p;
    };

    _Float16* wB1 = (_Float16*)carve((size_t)16 * 128 * 2);
    _Float16* wB2 = (_Float16*)carve((size_t)32 * 1024 * 2);
    _Float16* wB3 = (_Float16*)carve((size_t)32 * 512 * 2);
    _Float16* wBf = (_Float16*)carve((size_t)16 * 736 * 2);
    _Float16* xh  = (_Float16*)carve((size_t)4096 * 784 * 2);
    _Float16* a1  = (_Float16*)carve((size_t)4096 * 10 * 361 * 2);
    _Float16* a2  = (_Float16*)carve((size_t)4096 * 20 * 100 * 2);
    _Float16* a3  = (_Float16*)carve((size_t)4096 * 20 * 36 * 2);

    const int nX = 4096 * 784;
    pack_f32_to_f16<<<(nX + 255) / 256, 256, 0, stream>>>(x, xh, nX);

    prep_weights<<<(16 * 128  + 255) / 256, 256, 0, stream>>>(w1, wB1, 100,  128,  10, 16);
    prep_weights<<<(32 * 1024 + 255) / 256, 256, 0, stream>>>(w2, wB2, 1000, 1024, 20, 32);
    prep_weights<<<(32 * 512  + 255) / 256, 256, 0, stream>>>(w3, wB3, 500,  512,  20, 32);
    prep_weights<<<(16 * 736  + 255) / 256, 256, 0, stream>>>(wf, wBf, 720,  736,  10, 16);

    const int M1 = 4096 * 19 * 19;
    conv_gemm_wmma<_Float16, 1, 28, 28, 10, 10, 10, 19, 19, 100, 128, 16, true>
        <<<(M1 + 255) / 256, 256, 0, stream>>>(xh, wB1, b1, a1, M1);

    const int M2 = 4096 * 10 * 10;
    conv_gemm_wmma<_Float16, 10, 19, 19, 10, 10, 20, 10, 10, 1000, 1024, 32, true>
        <<<(M2 + 255) / 256, 256, 0, stream>>>(a1, wB2, b2, a2, M2);

    const int M3 = 4096 * 6 * 6;
    conv_gemm_wmma<_Float16, 20, 10, 10, 5, 5, 20, 6, 6, 500, 512, 32, true>
        <<<(M3 + 255) / 256, 256, 0, stream>>>(a2, wB3, b3, a3, M3);

    const int M4 = 4096;
    conv_gemm_wmma<float, 720, 1, 1, 1, 1, 10, 1, 1, 720, 736, 16, false>
        <<<(M4 + 255) / 256, 256, 0, stream>>>(a3, wBf, bf, out, M4);
}
